// pointnetbaseline_11433202942874
// MI455X (gfx1250) — matrix operand order repaired and checked by reading, not compiled, not run on hardware
//
#include <hip/hip_runtime.h>
#include <hip/hip_bf16.h>
#include <math.h>

// ---------------------------------------------------------------------------
// PointNet++ part-seg + MDN head, CDNA5 (gfx1250, wave32, WMMA).
// Activations are stored channel-major  A[c][p],  p = b*N + n  (f32).
// All 1x1-conv / FC layers run through one WMMA f16 GEMM kernel with fused
// BN+bias+ReLU epilogue; decoder fc3 fuses the 13-way output activations.
// ---------------------------------------------------------------------------

typedef __attribute__((ext_vector_type(16))) _Float16 v16h;
typedef __attribute__((ext_vector_type(8)))  float    v8f;

#define GEMM_BLOCK 256
#define GEMM_WAVES 8
#define LDS_KMAX   768               // max K over all layers (fp3 L1)
#define LDS_STRIDE (LDS_KMAX + 16)   // halves; multiple of 16 -> 32B aligned rows

// A-fragment K mapping (contiguous, matches B fragment):
// lane<16 : K=i
// lane>=16: K=i+16
__device__ __forceinline__ int a_kmap(int lane, int i) {
    bool lo = lane < 16;
    return lo ? i : i + 16;
}

// ---------------------------------------------------------------------------
// Weight prep
// ---------------------------------------------------------------------------
__global__ void fold_bn_kernel(const float* __restrict__ b,
                               const float* __restrict__ g,
                               const float* __restrict__ be,
                               const float* __restrict__ rm,
                               const float* __restrict__ rv,
                               float* __restrict__ scale,
                               float* __restrict__ bias2, int M) {
    int m = blockIdx.x * blockDim.x + threadIdx.x;
    if (m >= M) return;
    float s = g[m] * rsqrtf(rv[m] + 1e-5f);
    scale[m] = s;
    bias2[m] = (b[m] - rm[m]) * s + be[m];
}

// Pack W (M x K, row-major f32, optionally scaled by BN scale) into the WMMA
// A-fragment layout: Apack[(mt*KC+kc)*32 + lane] is a v16h.
__global__ void pack_w_kernel(const float* __restrict__ W,
                              const float* __restrict__ scale, // may be null
                              _Float16* __restrict__ Apack,
                              int M, int K, int MT, int KC) {
    int gid = blockIdx.x * blockDim.x + threadIdx.x;
    int tiles = MT * KC;
    if (gid >= tiles * 32) return;
    int lane = gid & 31;
    int tile = gid >> 5;
    int kc = tile % KC;
    int mt = tile / KC;
    int m = mt * 16 + (lane & 15);
    v16h v;
#pragma unroll
    for (int i = 0; i < 16; ++i) {
        int k = kc * 32 + a_kmap(lane, i);
        float f = 0.0f;
        if (m < M && k < K) {
            f = W[(size_t)m * K + k];
            if (scale) f *= scale[m];
        }
        v[i] = (_Float16)f;
    }
    ((v16h*)Apack)[(size_t)tile * 32 + lane] = v;
}

// ---------------------------------------------------------------------------
// WMMA GEMM:  Y[m][p] = act( sum_k Wpacked[m][k] * X[k][p] + bias[m] )
// act: 0 = none, 1 = relu, 2 = fused decoder-head epilogue.
// ---------------------------------------------------------------------------
__global__ __launch_bounds__(GEMM_BLOCK) void gemm_wmma(
    const _Float16* __restrict__ Apack,
    const float* __restrict__ X,
    const float* __restrict__ bias,
    float* __restrict__ Y,
    int M, int K, int KC, int P, int act,
    float* __restrict__ outFinal,   // act==2: d_out (13 x P x 20)
    float* __restrict__ piRaw)      // act==2: raw pi logits [20][P]
{
    __shared__ __align__(32) _Float16 sX[16 * LDS_STRIDE];
    const int tid  = threadIdx.x;
    const int n0   = blockIdx.x * 16;
    const int KC32 = KC * 32;

    // Stage X tile (K x 16) transposed into LDS as [n][k], zero-padded.
    for (int t = tid; t < 16 * KC32; t += GEMM_BLOCK) {
        int n = t & 15;
        int k = t >> 4;
        float v = (k < K) ? X[(size_t)k * P + (n0 + n)] : 0.0f;
        sX[n * LDS_STRIDE + k] = (_Float16)v;
    }
    __syncthreads();

    const int lane = tid & 31;
    const int wave = tid >> 5;
    const int mt   = blockIdx.y * GEMM_WAVES + wave;
    const int MT   = (M + 15) >> 4;
    if (mt >= MT) return;   // whole-wave uniform -> EXEC all-ones inside WMMA

    v8f acc;
#pragma unroll
    for (int r = 0; r < 8; ++r) acc[r] = 0.0f;

    const v16h* Af = (const v16h*)Apack;
    const int half = lane >> 4;
    // B fragment: col n = lane&15, halves i -> K = kc*32 + half*16 + i
    const _Float16* bbase = &sX[(lane & 15) * LDS_STRIDE + (half << 4)];

    for (int kc = 0; kc < KC; ++kc) {
        v16h a = Af[((size_t)mt * KC + kc) * 32 + lane];
        v16h b = *(const v16h*)(bbase + kc * 32);
        if (kc + 1 < KC)
            __builtin_prefetch(&Af[((size_t)mt * KC + kc + 1) * 32 + lane], 0, 0);
        acc = __builtin_amdgcn_wmma_f32_16x16x32_f16(
            false, a, false, b, (short)0, acc, false, false);
    }

    const int n     = n0 + (lane & 15);
    const int mbase = mt * 16 + (half << 3);

    if (act == 2) {
        const size_t P20 = (size_t)P * 20;
#pragma unroll
        for (int r = 0; r < 8; ++r) {
            int m = mbase + r;
            if (m >= M) continue;
            float v = acc[r] + bias[m];
            int t  = m / 20;
            int mm = m - t * 20;
            if (t == 0) {                         // pi: softmax over batch later
                piRaw[(size_t)mm * P + n] = v;
            } else {
                float o;
                if (t == 5 || t == 10)            o = tanhf(v);          // rho
                else if (t == 3 || t == 4 || t == 8 || t == 9 || t == 12)
                                                  o = expf(v);           // sigma
                else                              o = 1.0f / (1.0f + expf(-v)); // mu
                outFinal[(size_t)t * P20 + (size_t)n * 20 + mm] = o;
            }
        }
    } else {
#pragma unroll
        for (int r = 0; r < 8; ++r) {
            int m = mbase + r;
            if (m >= M) continue;
            float v = acc[r] + bias[m];
            if (act == 1) v = fmaxf(v, 0.0f);
            Y[(size_t)m * P + n] = v;
        }
    }
}

// ---------------------------------------------------------------------------
// Geometry / glue kernels
// ---------------------------------------------------------------------------
#define GID (blockIdx.x * blockDim.x + threadIdx.x)

// x (B,5,N) -> xyz0 [b][n][5] point-major  and xfeat [c][b*N+n] channel-major
__global__ void transpose_x_kernel(const float* __restrict__ x,
                                   float* __restrict__ xyz0,
                                   float* __restrict__ xfeat,
                                   int Bn, int N) {
    int g = GID;
    int total = Bn * 5 * N;
    if (g >= total) return;
    int n = g % N;
    int t = g / N;
    int c = t % 5;
    int b = t / 5;
    float v = x[g];
    xyz0[((size_t)b * N + n) * 5 + c] = v;
    xfeat[(size_t)c * Bn * N + (size_t)b * N + n] = v;
}

// Farthest point sampling (matches jax.lax.scan semantics: out[0]=0).
__global__ __launch_bounds__(256) void fps_kernel(const float* __restrict__ xyz,
                                                  float* __restrict__ dist,
                                                  int* __restrict__ out,
                                                  int N, int S) {
    __shared__ float rmax[256];
    __shared__ int   rarg[256];
    __shared__ int   sfar;
    int b = blockIdx.x, tid = threadIdx.x;
    const float* xb = xyz + (size_t)b * N * 5;
    float* db = dist + (size_t)b * N;
    for (int n = tid; n < N; n += 256) db[n] = 1e10f;
    if (tid == 0) { out[b * S] = 0; sfar = 0; }
    __syncthreads();
    for (int it = 1; it < S; ++it) {
        int far = sfar;
        float c0 = xb[far * 5 + 0], c1 = xb[far * 5 + 1], c2 = xb[far * 5 + 2];
        float c3 = xb[far * 5 + 3], c4 = xb[far * 5 + 4];
        float best = -1.0f; int barg = 0;
        for (int n = tid; n < N; n += 256) {
            const float* p = xb + (size_t)n * 5;
            float d0 = p[0] - c0, d1 = p[1] - c1, d2 = p[2] - c2;
            float d3 = p[3] - c3, d4 = p[4] - c4;
            float d = d0 * d0 + d1 * d1 + d2 * d2 + d3 * d3 + d4 * d4;
            float dn = db[n];
            dn = (d < dn) ? d : dn;
            db[n] = dn;
            if (dn > best) { best = dn; barg = n; }
        }
        rmax[tid] = best; rarg[tid] = barg;
        __syncthreads();
        for (int s = 128; s > 0; s >>= 1) {
            if (tid < s) {
                float av = rmax[tid], bv = rmax[tid + s];
                int ai = rarg[tid], bi = rarg[tid + s];
                if (bv > av || (bv == av && bi < ai)) { rmax[tid] = bv; rarg[tid] = bi; }
            }
            __syncthreads();
        }
        if (tid == 0) { sfar = rarg[0]; out[b * S + it] = rarg[0]; }
        __syncthreads();
    }
}

__global__ void gather_xyz_kernel(const float* __restrict__ xyz,
                                  const int* __restrict__ idx,
                                  float* __restrict__ out, int N, int S, int total) {
    int g = GID;
    if (g >= total) return;                 // total = B*S*5
    int c = g % 5;
    int t = g / 5;                          // t = b*S+s
    int b = t / S;
    out[g] = xyz[((size_t)b * N + idx[t]) * 5 + c];
}

// First-8-in-index-order ball query (== sort + take nsample in reference).
__global__ void ball_query_kernel(const float* __restrict__ xyz,
                                  const float* __restrict__ newxyz,
                                  int* __restrict__ gidx,
                                  int N, int S, float r2, int total) {
    int g = GID;
    if (g >= total) return;                 // total = B*S
    int b = g / S;
    float c0 = newxyz[(size_t)g * 5 + 0], c1 = newxyz[(size_t)g * 5 + 1];
    float c2 = newxyz[(size_t)g * 5 + 2], c3 = newxyz[(size_t)g * 5 + 3];
    float c4 = newxyz[(size_t)g * 5 + 4];
    int buf[8];
    int cnt = 0, first = -1;
    const float* xb = xyz + (size_t)b * N * 5;
    for (int n = 0; n < N; ++n) {
        const float* p = xb + (size_t)n * 5;
        float d0 = p[0] - c0, d1 = p[1] - c1, d2 = p[2] - c2;
        float d3 = p[3] - c3, d4 = p[4] - c4;
        float d = d0 * d0 + d1 * d1 + d2 * d2 + d3 * d3 + d4 * d4;
        if (d <= r2) {
            if (first < 0) first = n;
            buf[cnt++] = n;
            if (cnt == 8) break;
        }
    }
    if (first < 0) first = 0;
    for (int j = 0; j < 8; ++j) gidx[(size_t)g * 8 + j] = (j < cnt) ? buf[j] : first;
}

// Grouped features: rows 0..4 = centered xyz, rows 5.. = gathered features.
__global__ void group_kernel(const float* __restrict__ xyz,
                             const float* __restrict__ newxyz,
                             const int* __restrict__ gidx,
                             const float* __restrict__ Feat,
                             float* __restrict__ G,
                             int N, int S, int Cf, int FP, int P) {
    int g = GID;
    int C = 5 + Cf;
    if (g >= C * P) return;
    int c = g / P;
    int p = g % P;
    int bs = p >> 3;                        // nsample == 8
    int j = p & 7;
    int b = bs / S;
    int nn = gidx[(size_t)bs * 8 + j];
    float v;
    if (c < 5)
        v = xyz[((size_t)b * N + nn) * 5 + c] - newxyz[(size_t)bs * 5 + c];
    else
        v = Feat[(size_t)(c - 5) * FP + (size_t)b * N + nn];
    G[(size_t)c * P + p] = v;
}

__global__ void maxpool_kernel(const float* __restrict__ in,
                               float* __restrict__ out, int C, int Pout, int ns) {
    int g = GID;
    if (g >= C * Pout) return;
    int c = g / Pout;
    int q = g % Pout;
    const float* ip = in + (size_t)c * Pout * ns + (size_t)q * ns;
    float m = -3.0e38f;
    for (int j = 0; j < ns; ++j) m = fmaxf(m, ip[j]);
    out[(size_t)c * Pout + q] = m;
}

// sa3 group_all: rows 0..4 = l2 xyz (uncentered), rows 5..260 = l2 features.
__global__ void groupall_kernel(const float* __restrict__ nxyz2,
                                const float* __restrict__ l2f,
                                float* __restrict__ G, int P) {   // P = B*16
    int g = GID;
    if (g >= 261 * P) return;
    int c = g / P;
    int p = g % P;
    float v = (c < 5) ? nxyz2[(size_t)p * 5 + c]
                      : l2f[(size_t)(c - 5) * P + p];
    G[(size_t)c * P + p] = v;
}

// fp3: broadcast l3f [512][B] across the 16 l2 points, into rows [256..767].
__global__ void broadcast_l3_kernel(const float* __restrict__ l3f,
                                    float* __restrict__ dst, int P) {  // P = B*16
    int g = GID;
    if (g >= 512 * P) return;
    int c = g / P;
    int p = g % P;
    int b = p >> 4;
    dst[(size_t)c * P + p] = l3f[(size_t)c * 16 + b];
}

// 3-NN (matches top_k(-d,3): ascending d, earlier index wins ties).
__global__ void knn3_kernel(const float* __restrict__ q,
                            const float* __restrict__ s,
                            int* __restrict__ idx3, float* __restrict__ w3,
                            int NQ, int NS, int total) {
    int g = GID;
    if (g >= total) return;                 // total = B*NQ
    int b = g / NQ;
    const float* qp = q + (size_t)g * 5;
    float q0 = qp[0], q1 = qp[1], q2 = qp[2], q3 = qp[3], q4 = qp[4];
    float d0 = 3e38f, d1 = 3e38f, d2 = 3e38f;
    int i0 = 0, i1 = 0, i2 = 0;
    const float* sb = s + (size_t)b * NS * 5;
    for (int n = 0; n < NS; ++n) {
        const float* p = sb + (size_t)n * 5;
        float e0 = p[0] - q0, e1 = p[1] - q1, e2 = p[2] - q2;
        float e3 = p[3] - q3, e4 = p[4] - q4;
        float d = e0 * e0 + e1 * e1 + e2 * e2 + e3 * e3 + e4 * e4;
        if (d < d0) { d2 = d1; i2 = i1; d1 = d0; i1 = i0; d0 = d; i0 = n; }
        else if (d < d1) { d2 = d1; i2 = i1; d1 = d; i1 = n; }
        else if (d < d2) { d2 = d; i2 = n; }
    }
    float r0 = 1.0f / (d0 + 1e-8f), r1 = 1.0f / (d1 + 1e-8f), r2 = 1.0f / (d2 + 1e-8f);
    float inv = 1.0f / (r0 + r1 + r2);
    idx3[(size_t)g * 3 + 0] = i0; w3[(size_t)g * 3 + 0] = r0 * inv;
    idx3[(size_t)g * 3 + 1] = i1; w3[(size_t)g * 3 + 1] = r1 * inv;
    idx3[(size_t)g * 3 + 2] = i2; w3[(size_t)g * 3 + 2] = r2 * inv;
}

__global__ void interp3_kernel(const float* __restrict__ Feat,
                               const int* __restrict__ idx3,
                               const float* __restrict__ w3,
                               float* __restrict__ dst,
                               int C, int NQ, int NS, int PQ, int PS) {
    int g = GID;
    if (g >= C * PQ) return;
    int c = g / PQ;
    int p = g % PQ;
    int b = p / NQ;
    const int* ii = idx3 + (size_t)p * 3;
    const float* ww = w3 + (size_t)p * 3;
    size_t rb = (size_t)c * PS + (size_t)b * NS;
    dst[(size_t)c * PQ + p] =
        ww[0] * Feat[rb + ii[0]] + ww[1] * Feat[rb + ii[1]] + ww[2] * Feat[rb + ii[2]];
}

__global__ void logsoftmax_kernel(const float* __restrict__ in,
                                  float* __restrict__ out, int C, int P) {
    int p = GID;
    if (p >= P) return;
    float mx = -3e38f;
    for (int c = 0; c < C; ++c) mx = fmaxf(mx, in[(size_t)c * P + p]);
    float s = 0.0f;
    for (int c = 0; c < C; ++c) s += expf(in[(size_t)c * P + p] - mx);
    float ls = mx + logf(s);
    for (int c = 0; c < C; ++c) out[(size_t)c * P + p] = in[(size_t)c * P + p] - ls;
}

// pi = softmax over the batch axis (axis 0 of (B,N,20)).
__global__ void pi_softmax_kernel(const float* __restrict__ piRaw,
                                  float* __restrict__ out,
                                  int Bn, int Npts) {
    int g = GID;
    if (g >= Npts * 20) return;
    int n = g / 20;
    int m = g % 20;
    size_t P = (size_t)Bn * Npts;
    float mx = -3e38f;
    for (int b = 0; b < 16; ++b)
        mx = fmaxf(mx, piRaw[(size_t)m * P + (size_t)b * Npts + n]);
    float e[16], s = 0.0f;
    for (int b = 0; b < 16; ++b) {
        e[b] = expf(piRaw[(size_t)m * P + (size_t)b * Npts + n] - mx);
        s += e[b];
    }
    float inv = 1.0f / s;
    for (int b = 0; b < 16; ++b)
        out[((size_t)b * Npts + n) * 20 + m] = e[b] * inv;
}

// ---------------------------------------------------------------------------
// Host orchestration
// ---------------------------------------------------------------------------
struct LSpec { int w, b, bn, M, K; };  // bn = index of gamma (or -1)

extern "C" void kernel_launch(void* const* d_in, const int* in_sizes, int n_in,
                              void* d_out, int out_size, void* d_ws, size_t ws_size,
                              hipStream_t stream) {
    (void)in_sizes; (void)n_in; (void)out_size; (void)ws_size;

    const int Bn = 16, Npts = 16384;
    const int Ptot = Bn * Npts;           // 262144
    const int S1 = 32, S2 = 16, NS = 8;

    auto Fin = [&](int i) -> const float* { return (const float*)d_in[i]; };

    // Input leaf order (insertion-order pytree flatten of setup_inputs()).
    static const LSpec LS[23] = {
        {  1,   2,   7,  64,  10}, {  3,   4,  11,  64,  64}, {  5,   6,  15, 128,  64},   // sa1
        { 19,  20,  25, 128, 133}, { 21,  22,  29, 128, 128}, { 23,  24,  33, 256, 128},   // sa2
        { 37,  38,  43, 256, 261}, { 39,  40,  47, 256, 256}, { 41,  42,  51, 512, 256},   // sa3
        { 55,  56,  59, 256, 768}, { 57,  58,  63, 256, 256},                              // fp3
        { 67,  68,  71, 256, 384}, { 69,  70,  75, 128, 256},                              // fp2
        { 79,  80,  85, 128, 128}, { 81,  82,  89, 128, 128}, { 83,  84,  93, 128, 128},   // fp1
        { 97,  98, 105, 100, 128}, { 99, 100, 109,  50, 100}, {101, 102, 113,  25,  50},
        {103, 104,  -1, 128,  25},                                                          // head
        {117, 118,  -1, 128, 128}, {119, 120,  -1,  64, 128}, {121, 122,  -1, 260,  64}    // dec
    };

    // Bump allocator over d_ws.
    char* base = (char*)d_ws;
    size_t cur = 0;
    auto alloc = [&](size_t bytes) -> void* {
        size_t a = (cur + 255) & ~(size_t)255;
        cur = a + bytes;
        return base + a;
    };

    float* xyz0  = (float*)alloc((size_t)Ptot * 5 * 4);
    float* xfeat = (float*)alloc((size_t)5 * Ptot * 4);
    float* fdist = (float*)alloc((size_t)Ptot * 4);
    int*   fidx1 = (int*)alloc((size_t)Bn * S1 * 4);
    float* nxyz1 = (float*)alloc((size_t)Bn * S1 * 5 * 4);
    int*   gidx1 = (int*)alloc((size_t)Bn * S1 * NS * 4);
    int*   fidx2 = (int*)alloc((size_t)Bn * S2 * 4);
    float* nxyz2 = (float*)alloc((size_t)Bn * S2 * 5 * 4);
    int*   gidx2 = (int*)alloc((size_t)Bn * S2 * NS * 4);
    float* l1f   = (float*)alloc((size_t)128 * Bn * S1 * 4);
    float* l2f   = (float*)alloc((size_t)256 * Bn * S2 * 4);
    float* l3f   = (float*)alloc((size_t)512 * Bn * 4);
    float* l2f2  = (float*)alloc((size_t)256 * Bn * S2 * 4);
    float* l1f2  = (float*)alloc((size_t)128 * Bn * S1 * 4);
    int*   idx3a = (int*)alloc((size_t)Bn * S1 * 3 * 4);
    float* w3a   = (float*)alloc((size_t)Bn * S1 * 3 * 4);
    int*   idx3b = (int*)alloc((size_t)Ptot * 3 * 4);
    float* w3b   = (float*)alloc((size_t)Ptot * 3 * 4);
    float* SA    = (float*)alloc((size_t)768 * 4096 * 4);
    float* SB    = (float*)alloc((size_t)768 * 4096 * 4);
    float* piRaw = (float*)alloc((size_t)20 * Ptot * 4);
    float* BigA  = (float*)alloc((size_t)128 * Ptot * 4);
    float* BigB  = (float*)alloc((size_t)128 * Ptot * 4);

    // --- weight prep: fold BN, pack WMMA A-fragments --------------------------
    struct LDev { const _Float16* Apack; const float* bias; int M, K, KC, MT; };
    LDev LD[23];
    for (int i = 0; i < 23; ++i) {
        int M = LS[i].M, K = LS[i].K;
        int MT = (M + 15) / 16, KC = (K + 31) / 32;
        _Float16* Apack = (_Float16*)alloc((size_t)MT * KC * 32 * 16 * 2);
        const float* scale = nullptr;
        const float* bias;
        if (LS[i].bn >= 0) {
            float* sc = (float*)alloc((size_t)M * 4);
            float* b2 = (float*)alloc((size_t)M * 4);
            fold_bn_kernel<<<dim3((M + 127) / 128), dim3(128), 0, stream>>>(
                Fin(LS[i].b), Fin(LS[i].bn), Fin(LS[i].bn + 1),
                Fin(LS[i].bn + 2), Fin(LS[i].bn + 3), sc, b2, M);
            scale = sc; bias = b2;
        } else {
            bias = Fin(LS[i].b);
        }
        int tot = MT * KC * 32;
        pack_w_kernel<<<dim3((tot + 127) / 128), dim3(128), 0, stream>>>(
            Fin(LS[i].w), scale, Apack, M, K, MT, KC);
        LD[i] = {Apack, bias, M, K, KC, MT};
    }

    auto gemm = [&](int li, const float* X, float* Y, int P, int act,
                    float* outF, float* piR) {
        dim3 grid(P / 16, (LD[li].MT + GEMM_WAVES - 1) / GEMM_WAVES);
        gemm_wmma<<<grid, dim3(GEMM_BLOCK), 0, stream>>>(
            LD[li].Apack, X, LD[li].bias, Y,
            LD[li].M, LD[li].K, LD[li].KC, P, act, outF, piR);
    };
    auto launch = [&](int total) { return dim3((total + 255) / 256); };

    // --- input transpose -----------------------------------------------------
    transpose_x_kernel<<<launch(Bn * 5 * Npts), dim3(256), 0, stream>>>(
        (const float*)d_in[0], xyz0, xfeat, Bn, Npts);

    // --- sa1 -----------------------------------------------------------------
    fps_kernel<<<dim3(Bn), dim3(256), 0, stream>>>(xyz0, fdist, fidx1, Npts, S1);
    gather_xyz_kernel<<<launch(Bn * S1 * 5), dim3(256), 0, stream>>>(
        xyz0, fidx1, nxyz1, Npts, S1, Bn * S1 * 5);
    ball_query_kernel<<<launch(Bn * S1), dim3(256), 0, stream>>>(
        xyz0, nxyz1, gidx1, Npts, S1, 0.04f, Bn * S1);
    int P1 = Bn * S1 * NS;   // 4096
    group_kernel<<<launch(10 * P1), dim3(256), 0, stream>>>(
        xyz0, nxyz1, gidx1, xfeat, SA, Npts, S1, 5, Ptot, P1);
    gemm(0, SA, SB, P1, 1, nullptr, nullptr);
    gemm(1, SB, SA, P1, 1, nullptr, nullptr);
    gemm(2, SA, SB, P1, 1, nullptr, nullptr);
    maxpool_kernel<<<launch(128 * Bn * S1), dim3(256), 0, stream>>>(SB, l1f, 128, Bn * S1, NS);

    // --- sa2 -----------------------------------------------------------------
    fps_kernel<<<dim3(Bn), dim3(256), 0, stream>>>(nxyz1, fdist, fidx2, S1, S2);
    gather_xyz_kernel<<<launch(Bn * S2 * 5), dim3(256), 0, stream>>>(
        nxyz1, fidx2, nxyz2, S1, S2, Bn * S2 * 5);
    ball_query_kernel<<<launch(Bn * S2), dim3(256), 0, stream>>>(
        nxyz1, nxyz2, gidx2, S1, S2, 0.16f, Bn * S2);
    int P2 = Bn * S2 * NS;   // 2048
    group_kernel<<<launch(133 * P2), dim3(256), 0, stream>>>(
        nxyz1, nxyz2, gidx2, l1f, SA, S1, S2, 128, Bn * S1, P2);
    gemm(3, SA, SB, P2, 1, nullptr, nullptr);
    gemm(4, SB, SA, P2, 1, nullptr, nullptr);
    gemm(5, SA, SB, P2, 1, nullptr, nullptr);
    maxpool_kernel<<<launch(256 * Bn * S2), dim3(256), 0, stream>>>(SB, l2f, 256, Bn * S2, NS);

    // --- sa3 (group_all) -----------------------------------------------------
    int P3 = Bn * S2;        // 256
    groupall_kernel<<<launch(261 * P3), dim3(256), 0, stream>>>(nxyz2, l2f, SA, P3);
    gemm(6, SA, SB, P3, 1, nullptr, nullptr);
    gemm(7, SB, SA, P3, 1, nullptr, nullptr);
    gemm(8, SA, SB, P3, 1, nullptr, nullptr);
    maxpool_kernel<<<launch(512 * Bn), dim3(256), 0, stream>>>(SB, l3f, 512, Bn, 16);

    // --- fp3 -----------------------------------------------------------------
    hipMemcpyAsync(SA, l2f, (size_t)256 * P3 * 4, hipMemcpyDeviceToDevice, stream);
    broadcast_l3_kernel<<<launch(512 * P3), dim3(256), 0, stream>>>(
        l3f, SA + (size_t)256 * P3, P3);
    gemm(9, SA, SB, P3, 1, nullptr, nullptr);
    gemm(10, SB, SA, P3, 1, nullptr, nullptr);
    hipMemcpyAsync(l2f2, SA, (size_t)256 * P3 * 4, hipMemcpyDeviceToDevice, stream);

    // --- fp2 -----------------------------------------------------------------
    knn3_kernel<<<launch(Bn * S1), dim3(256), 0, stream>>>(
        nxyz1, nxyz2, idx3a, w3a, S1, S2, Bn * S1);
    hipMemcpyAsync(SA, l1f, (size_t)128 * Bn * S1 * 4, hipMemcpyDeviceToDevice, stream);
    interp3_kernel<<<launch(256 * Bn * S1), dim3(256), 0, stream>>>(
        l2f2, idx3a, w3a, SA + (size_t)128 * Bn * S1, 256, S1, S2, Bn * S1, Bn * S2);
    gemm(11, SA, SB, Bn * S1, 1, nullptr, nullptr);
    gemm(12, SB, SA, Bn * S1, 1, nullptr, nullptr);
    hipMemcpyAsync(l1f2, SA, (size_t)128 * Bn * S1 * 4, hipMemcpyDeviceToDevice, stream);

    // --- fp1 (dominant stage, P = 262144) ------------------------------------
    knn3_kernel<<<launch(Ptot), dim3(256), 0, stream>>>(
        xyz0, nxyz1, idx3b, w3b, Npts, S1, Ptot);
    interp3_kernel<<<launch(128 * Ptot), dim3(256), 0, stream>>>(
        l1f2, idx3b, w3b, BigA, 128, Npts, S1, Ptot, Bn * S1);
    gemm(13, BigA, BigB, Ptot, 1, nullptr, nullptr);
    gemm(14, BigB, BigA, Ptot, 1, nullptr, nullptr);
    gemm(15, BigA, BigB, Ptot, 1, nullptr, nullptr);

    // --- head ----------------------------------------------------------------
    gemm(16, BigB, BigA, Ptot, 1, nullptr, nullptr);
    gemm(17, BigA, BigB, Ptot, 1, nullptr, nullptr);
    gemm(18, BigB, BigA, Ptot, 1, nullptr, nullptr);
    gemm(19, BigA, BigB, Ptot, 0, nullptr, nullptr);   // conv only, no bn/relu
    logsoftmax_kernel<<<launch(Ptot), dim3(256), 0, stream>>>(BigB, BigA, 128, Ptot);

    // --- decoder (fc3 epilogue fuses the 13-way output activations) ----------
    gemm(20, BigA, BigB, Ptot, 1, nullptr, nullptr);
    gemm(21, BigB, BigA, Ptot, 1, nullptr, nullptr);
    gemm(22, BigA, nullptr, Ptot, 2, (float*)d_out, piRaw);
    pi_softmax_kernel<<<launch(Npts * 20), dim3(256), 0, stream>>>(
        piRaw, (float*)d_out, Bn, Npts);
}
